// Multiplication_GRU_18227841204355
// MI455X (gfx1250) — matrix operand order repaired and checked by reading, not compiled, not run on hardware
//
#include <hip/hip_runtime.h>

typedef __attribute__((ext_vector_type(16))) _Float16 v16h;
typedef __attribute__((ext_vector_type(8)))  float    v8f;

#define Bsz   512
#define Tsz   1024
#define Hsz   128
#define MT    16          // batches per block (WMMA M)
#define GH_STRIDE 388     // padded 384 to avoid LDS bank aliasing
#define NT_PER_WAVE 3     // 24 N-tiles (384/16) over 8 waves

__global__ __launch_bounds__(256)
void gru_fused_kernel(const float* __restrict__ x, const int* __restrict__ lengths,
                      const float* __restrict__ w_ih, const float* __restrict__ w_hh,
                      const float* __restrict__ b_ih, const float* __restrict__ b_hh,
                      const float* __restrict__ head_w, const float* __restrict__ head_b,
                      float* __restrict__ out)
{
    __shared__ float ghL[MT * GH_STRIDE];            // gh staging [16][388] f32
    __shared__ float hL[MT * Hsz];                   // h fp32
    __shared__ __align__(32) _Float16 hA[4 * 512];   // h f16 in WMMA A-fragment layout (4 K-tiles)
    __shared__ float xtile[MT][2];
    __shared__ int   lensL[MT];

    const int tid  = threadIdx.x;
    const int lane = tid & 31;
    const int wave = tid >> 5;
    const int m0   = blockIdx.x * MT;

    // ---- w_hh B-matrix fragments, register resident for whole kernel ----
    // B[K][N] = w_hh[N][K];  lane: N = lane&15, K base = (lane>>4)*8; element e -> K offset (e&7) + (e>>3)*16
    v16h bfrag[NT_PER_WAVE * 4];
    {
        const int ncol  = lane & 15;
        const int khalf = (lane >> 4) * 8;
#pragma unroll
        for (int j = 0; j < NT_PER_WAVE; ++j) {
            const int N = (wave * NT_PER_WAVE + j) * 16 + ncol;
#pragma unroll
            for (int kt = 0; kt < 4; ++kt) {
                const float* src = w_hh + N * Hsz + kt * 32 + khalf;
                v16h f;
#pragma unroll
                for (int e = 0; e < 16; ++e) f[e] = (_Float16)src[(e & 7) + (e >> 3) * 16];
                bfrag[j * 4 + kt] = f;
            }
        }
    }

    // ---- per-thread gate constants: thread owns hidden column k, 8 batches ----
    const int k  = tid & (Hsz - 1);
    const int bg = tid >> 7;                         // 0 or 1
    const float wr0 = w_ih[k*2+0],            wr1 = w_ih[k*2+1];
    const float wz0 = w_ih[(Hsz+k)*2+0],      wz1 = w_ih[(Hsz+k)*2+1];
    const float wn0 = w_ih[(2*Hsz+k)*2+0],    wn1 = w_ih[(2*Hsz+k)*2+1];
    const float br  = b_ih[k]       + b_hh[k];
    const float bz  = b_ih[Hsz+k]   + b_hh[Hsz+k];
    const float bin = b_ih[2*Hsz+k];                 // stays outside r*hn
    const float bhn = b_hh[2*Hsz+k];                 // inside r*hn
    const float hw  = head_w[k];

    // hA scatter address for (b, k) per documented 16-bit A 16x32 layout
    const int kt_    = k >> 5, klocal = k & 31;
    const int grp    = klocal >> 4;
    const int hh     = (klocal >> 3) & 1;
    const int e_     = grp * 8 + (klocal & 7);
    const int scat0  = kt_ * 512 + hh * 256 + e_;    // + b*16

    for (int i = tid; i < MT * Hsz; i += 256) hL[i] = 0.0f;
    for (int i = tid; i < 4 * 512;  i += 256) hA[i] = (_Float16)0.0f;
    if (tid < MT) lensL[tid] = lengths[m0 + tid];
    __syncthreads();

    int maxlen = 0;
    int lenr[8];
#pragma unroll
    for (int i = 0; i < MT; ++i) { int l = lensL[i]; maxlen = l > maxlen ? l : maxlen; }
#pragma unroll
    for (int i = 0; i < 8; ++i) lenr[i] = lensL[bg * 8 + i];

    float part[8];
#pragma unroll
    for (int i = 0; i < 8; ++i) part[i] = 0.0f;

    const int ncolC = lane & 15;
    const int mrow  = (lane >> 4) * 8;               // C-layout: M = vgpr + 8*(lane/16)
    const int n0    = (wave * NT_PER_WAVE + 0) * 16 + ncolC;
    const int n1    = (wave * NT_PER_WAVE + 1) * 16 + ncolC;
    const int n2    = (wave * NT_PER_WAVE + 2) * 16 + ncolC;

    for (int t = 0; t < maxlen; ++t) {
        // prefetch x(t) for this batch tile (hidden under WMMA phase)
        float xv0 = 0.0f, xv1 = 0.0f;
        if (tid < MT) {
            const float* xp = x + ((size_t)(m0 + tid) * Tsz + t) * 2;
            xv0 = xp[0]; xv1 = xp[1];
        }

        // ---- Phase A: gh = h @ w_hh^T, 12 WMMAs per wave, 3 chains interleaved ----
        v16h a0 = *(const v16h*)&hA[0*512 + lane*16];
        v16h a1 = *(const v16h*)&hA[1*512 + lane*16];
        v16h a2 = *(const v16h*)&hA[2*512 + lane*16];
        v16h a3 = *(const v16h*)&hA[3*512 + lane*16];

        v8f acc0 = {}, acc1 = {}, acc2 = {};
        acc0 = __builtin_amdgcn_wmma_f32_16x16x32_f16(false, a0, false, bfrag[0], (short)0, acc0, false, false);
        acc1 = __builtin_amdgcn_wmma_f32_16x16x32_f16(false, a0, false, bfrag[4], (short)0, acc1, false, false);
        acc2 = __builtin_amdgcn_wmma_f32_16x16x32_f16(false, a0, false, bfrag[8], (short)0, acc2, false, false);
        acc0 = __builtin_amdgcn_wmma_f32_16x16x32_f16(false, a1, false, bfrag[1], (short)0, acc0, false, false);
        acc1 = __builtin_amdgcn_wmma_f32_16x16x32_f16(false, a1, false, bfrag[5], (short)0, acc1, false, false);
        acc2 = __builtin_amdgcn_wmma_f32_16x16x32_f16(false, a1, false, bfrag[9], (short)0, acc2, false, false);
        acc0 = __builtin_amdgcn_wmma_f32_16x16x32_f16(false, a2, false, bfrag[2], (short)0, acc0, false, false);
        acc1 = __builtin_amdgcn_wmma_f32_16x16x32_f16(false, a2, false, bfrag[6], (short)0, acc1, false, false);
        acc2 = __builtin_amdgcn_wmma_f32_16x16x32_f16(false, a2, false, bfrag[10], (short)0, acc2, false, false);
        acc0 = __builtin_amdgcn_wmma_f32_16x16x32_f16(false, a3, false, bfrag[3], (short)0, acc0, false, false);
        acc1 = __builtin_amdgcn_wmma_f32_16x16x32_f16(false, a3, false, bfrag[7], (short)0, acc1, false, false);
        acc2 = __builtin_amdgcn_wmma_f32_16x16x32_f16(false, a3, false, bfrag[11], (short)0, acc2, false, false);

#pragma unroll
        for (int r = 0; r < 8; ++r) {
            ghL[(mrow + r) * GH_STRIDE + n0] = acc0[r];
            ghL[(mrow + r) * GH_STRIDE + n1] = acc1[r];
            ghL[(mrow + r) * GH_STRIDE + n2] = acc2[r];
        }

        if (tid < MT) { xtile[tid][0] = xv0; xtile[tid][1] = xv1; }
        __syncthreads();

        // ---- Phase B: gates + h update (8 batches per thread) ----
#pragma unroll
        for (int i = 0; i < 8; ++i) {
            const int b  = bg * 8 + i;
            const float x0 = xtile[b][0], x1 = xtile[b][1];
            const float hr = ghL[b * GH_STRIDE + k];
            const float hz = ghL[b * GH_STRIDE + Hsz + k];
            const float hn = ghL[b * GH_STRIDE + 2*Hsz + k] + bhn;
            const float ar = wr0*x0 + wr1*x1 + br + hr;
            const float az = wz0*x0 + wz1*x1 + bz + hz;
            const float an = wn0*x0 + wn1*x1 + bin;
            const float r  = 1.0f / (1.0f + __expf(-ar));
            const float z  = 1.0f / (1.0f + __expf(-az));
            const float na = an + r * hn;
            const float et = __expf(-2.0f * na);
            const float n  = (1.0f - et) / (1.0f + et);
            const float h  = hL[b * Hsz + k];
            const float hnew = (1.0f - z) * n + z * h;
            hL[b * Hsz + k] = hnew;
            hA[scat0 + b * 16] = (_Float16)hnew;      // pre-swizzled A fragment
            if (t == lenr[i] - 1) part[i] = hw * hnew; // fused gather + head dot
        }
        __syncthreads();
    }

    // ---- head reduction: out[b] = sum_k head_w[k]*h_last[b][k] + head_b ----
#pragma unroll
    for (int i = 0; i < 8; ++i)
        ghL[(bg * 8 + i) * GH_STRIDE + k] = part[i];
    __syncthreads();
    if (tid < MT) {
        float s = 0.0f;
        for (int kk = 0; kk < Hsz; ++kk) s += ghL[tid * GH_STRIDE + kk];
        out[m0 + tid] = s + head_b[0];
    }
}

extern "C" void kernel_launch(void* const* d_in, const int* in_sizes, int n_in,
                              void* d_out, int out_size, void* d_ws, size_t ws_size,
                              hipStream_t stream) {
    (void)in_sizes; (void)n_in; (void)d_ws; (void)ws_size; (void)out_size;
    const float* x      = (const float*)d_in[0];
    const int*   len    = (const int*)  d_in[1];
    const float* w_ih   = (const float*)d_in[2];
    const float* w_hh   = (const float*)d_in[3];
    const float* b_ih   = (const float*)d_in[4];
    const float* b_hh   = (const float*)d_in[5];
    const float* head_w = (const float*)d_in[6];
    const float* head_b = (const float*)d_in[7];
    float* out = (float*)d_out;

    dim3 grid(Bsz / MT);   // 32 workgroups of 16 batches
    dim3 block(256);       // 8 wave32
    gru_fused_kernel<<<grid, block, 0, stream>>>(x, len, w_ih, w_hh, b_ih, b_hh,
                                                 head_w, head_b, out);
}
